// GCN_30107720745357
// MI455X (gfx1250) — matrix operand order repaired and checked by reading, not compiled, not run on hardware
//
#include <hip/hip_runtime.h>
#include <cstdint>
#include <cstddef>

#define NN 100000
#define NE 1600000
#define NFEAT 128
#define NHID 128
#define NCLASS 64

typedef __attribute__((ext_vector_type(16))) __bf16 v16bf;
typedef __attribute__((ext_vector_type(8)))  float  v8f;

// ---------------- graph preprocessing ----------------

__global__ __launch_bounds__(256) void k_zero2(int* __restrict__ a, int* __restrict__ b, int n) {
    int i = blockIdx.x * 256 + threadIdx.x;
    if (i < n) { a[i] = 0; b[i] = 0; }
}

__global__ __launch_bounds__(256) void k_count(const int* __restrict__ rows, int* __restrict__ cnt, int e) {
    int i = blockIdx.x * 256 + threadIdx.x;
    if (i < e) atomicAdd(&cnt[rows[i]], 1);
}

__global__ __launch_bounds__(256) void k_dinv(const int* __restrict__ cnt, float* __restrict__ dinv, int n) {
    int i = blockIdx.x * 256 + threadIdx.x;
    if (i < n) dinv[i] = rsqrtf((float)(cnt[i] + 1));   // +1 self-loop; deg>0 always
}

__global__ __launch_bounds__(256) void k_scan_local(const int* __restrict__ cnt, int* __restrict__ rowptr,
                                                    int* __restrict__ bsums, int n) {
    __shared__ int s[256];
    const int tid = threadIdx.x;
    const int idx = blockIdx.x * 256 + tid;
    int v = (idx < n) ? cnt[idx] : 0;
    s[tid] = v;
    __syncthreads();
    for (int off = 1; off < 256; off <<= 1) {
        int t = (tid >= off) ? s[tid - off] : 0;
        __syncthreads();
        s[tid] += t;
        __syncthreads();
    }
    if (idx < n) rowptr[idx] = s[tid] - v;   // exclusive within block
    if (tid == 255) bsums[blockIdx.x] = s[255];
}

__global__ void k_scan_sums(int* __restrict__ bsums, int nb, int* __restrict__ rowptr, int n) {
    if (threadIdx.x == 0 && blockIdx.x == 0) {
        int run = 0;
        for (int b = 0; b < nb; ++b) { int t = bsums[b]; bsums[b] = run; run += t; }
        rowptr[n] = run;
    }
}

__global__ __launch_bounds__(256) void k_scan_add(int* __restrict__ rowptr, const int* __restrict__ bsums, int n) {
    int i = blockIdx.x * 256 + threadIdx.x;
    if (i < n) rowptr[i] += bsums[i >> 8];
}

__global__ __launch_bounds__(256) void k_fill(const int* __restrict__ rows, const int* __restrict__ colsIn,
                                              const float* __restrict__ dinv, const int* __restrict__ rowptr,
                                              int* __restrict__ fillc, int* __restrict__ ccol,
                                              float* __restrict__ cw, int e) {
    int i = blockIdx.x * 256 + threadIdx.x;
    if (i >= e) return;
    const int r = rows[i];
    const int c = colsIn[i];
    const int pos = rowptr[r] + atomicAdd(&fillc[r], 1);
    ccol[pos] = c;
    cw[pos] = dinv[r] * dinv[c];     // D^-1/2 A D^-1/2
}

// Transposed bf16 weights: wt[n*K + k] = (bf16)W[k*ncol + n]
__global__ __launch_bounds__(256) void k_wconv(const float* __restrict__ W, __bf16* __restrict__ wt,
                                               int K, int ncol) {
    int idx = blockIdx.x * 256 + threadIdx.x;
    if (idx >= K * ncol) return;
    int n = idx / K, k = idx % K;
    wt[(size_t)n * K + k] = (__bf16)W[(size_t)k * ncol + n];
}

// ---------------- bf16 WMMA GEMM: dst[N x DOUT] = src[N x 128] @ W ----------------
// One wave per 16-row stripe; NT = DOUT/16 column tiles; K=128 in 4 steps of 32.
// B panel staged once per block into padded LDS (row stride K+16 bf16 = 288B:
// keeps 32B alignment for v16bf ds-loads, 8-bank phase shift per column kills
// the 16-way bank conflict of a 256B stride).
// A frag (16x32 bf16, contiguous k order shared with B): lane half h = lane>>4, row m = m0 + (lane&15);
//   elems j = src[m][k0 + 16h + j]
// B frag (32x16): col n = t*16 + (lane&15); elems j = W[k0 + 16h + j][n]  -> contiguous in LDS.
template<int DOUT>
__global__ __launch_bounds__(256) void k_gemm(const float* __restrict__ src,
                                              const __bf16* __restrict__ wt,
                                              float* __restrict__ dst, int nrows) {
    constexpr int K  = 128;
    constexpr int KP = K + 16;             // padded LDS row stride in bf16 units
    constexpr int NT = DOUT / 16;
    __shared__ __bf16 ldsW[DOUT * KP];     // 36.9KB (DOUT=128) / 18.4KB (DOUT=64)

    // cooperative stage: 16B per thread per step, coalesced reads from wt
    constexpr int CH = (DOUT * K) / 8;     // 8-element (16B) chunks
    for (int idx = threadIdx.x; idx < CH; idx += 256) {
        const int n  = idx >> 4;           // K/8 = 16 chunks per column
        const int kk = (idx & 15) * 8;
        *(float4*)(&ldsW[n * KP + kk]) = *(const float4*)(wt + (size_t)n * K + kk);
    }
    __syncthreads();

    const int lane = threadIdx.x & 31;
    const int wave = threadIdx.x >> 5;
    const int half = lane >> 4;
    const int l16  = lane & 15;
    const int m0 = (blockIdx.x * 8 + wave) * 16;
    if (m0 >= nrows) return;               // wave-uniform: EXEC all-ones for WMMA

    const v8f vzero = {0.f, 0.f, 0.f, 0.f, 0.f, 0.f, 0.f, 0.f};
    v8f acc[NT];
#pragma unroll
    for (int t = 0; t < NT; ++t) acc[t] = vzero;

    const float* arow = src + (size_t)(m0 + l16) * K;
#pragma unroll
    for (int k0 = 0; k0 < K; k0 += 32) {
        const float4 q0 = *(const float4*)(arow + k0 + half * 16);
        const float4 q1 = *(const float4*)(arow + k0 + half * 16 + 4);
        const float4 q2 = *(const float4*)(arow + k0 + half * 16 + 8);
        const float4 q3 = *(const float4*)(arow + k0 + half * 16 + 12);
        v16bf a;
        a[0]  = (__bf16)q0.x; a[1]  = (__bf16)q0.y; a[2]  = (__bf16)q0.z; a[3]  = (__bf16)q0.w;
        a[4]  = (__bf16)q1.x; a[5]  = (__bf16)q1.y; a[6]  = (__bf16)q1.z; a[7]  = (__bf16)q1.w;
        a[8]  = (__bf16)q2.x; a[9]  = (__bf16)q2.y; a[10] = (__bf16)q2.z; a[11] = (__bf16)q2.w;
        a[12] = (__bf16)q3.x; a[13] = (__bf16)q3.y; a[14] = (__bf16)q3.z; a[15] = (__bf16)q3.w;
#pragma unroll
        for (int t = 0; t < NT; ++t) {
            const v16bf b = *(const v16bf*)(&ldsW[(t * 16 + l16) * KP + k0 + half * 16]);
            acc[t] = __builtin_amdgcn_wmma_f32_16x16x32_bf16(
                false, a, false, b, (short)0, acc[t], false, false);
        }
    }
    // D layout: VGPR v -> row m0 + half*8 + v, col = t*16 + l16
#pragma unroll
    for (int t = 0; t < NT; ++t) {
        const int n = t * 16 + l16;
#pragma unroll
        for (int v = 0; v < 8; ++v) {
            dst[(size_t)(m0 + half * 8 + v) * DOUT + n] = acc[t][v];
        }
    }
}

// ---------------- CSR gather aggregation: out = A_norm @ tmp + b (opt ReLU) ----------------

__device__ __forceinline__ void vload(const float* p, float (&v)[4]) {
    const float4 t = *(const float4*)p; v[0] = t.x; v[1] = t.y; v[2] = t.z; v[3] = t.w;
}
__device__ __forceinline__ void vload(const float* p, float (&v)[2]) {
    const float2 t = *(const float2*)p; v[0] = t.x; v[1] = t.y;
}

template<int DCOL, bool RELU>
__global__ __launch_bounds__(256) void k_agg(const float* __restrict__ tmp,
                                             const int* __restrict__ rowptr,
                                             const int* __restrict__ ccol,
                                             const float* __restrict__ cw,
                                             const float* __restrict__ dinv,
                                             const float* __restrict__ bias,
                                             float* __restrict__ outp, int n) {
    constexpr int V = DCOL / 32;                  // floats per lane (4 or 2)
    const int lane = threadIdx.x & 31;
    const int node = blockIdx.x * 8 + (threadIdx.x >> 5);   // one wave32 per node
    if (node >= n) return;

    const float di = dinv[node];
    float acc[V], t[V];
    vload(tmp + (size_t)node * DCOL + lane * V, t);
#pragma unroll
    for (int i = 0; i < V; ++i) acc[i] = di * di * t[i];    // self-loop weight dinv^2

    const int end = rowptr[node + 1];
#pragma unroll 2
    for (int j = rowptr[node]; j < end; ++j) {
        const int   c = ccol[j];
        const float w = cw[j];
        vload(tmp + (size_t)c * DCOL + lane * V, t);        // 512B/256B row gather, L2-resident
#pragma unroll
        for (int i = 0; i < V; ++i) acc[i] = fmaf(w, t[i], acc[i]);
    }
#pragma unroll
    for (int i = 0; i < V; ++i) {
        float r = acc[i] + bias[lane * V + i];
        if (RELU) r = fmaxf(r, 0.f);
        outp[(size_t)node * DCOL + lane * V + i] = r;
    }
}

// ---------------- launcher ----------------

extern "C" void kernel_launch(void* const* d_in, const int* in_sizes, int n_in,
                              void* d_out, int out_size, void* d_ws, size_t ws_size,
                              hipStream_t stream) {
    const float* x  = (const float*)d_in[0];
    const int*   ei = (const int*)  d_in[1];
    const float* W0 = (const float*)d_in[2];
    const float* b0 = (const float*)d_in[3];
    const float* W1 = (const float*)d_in[4];
    const float* b1 = (const float*)d_in[5];
    const float* W2 = (const float*)d_in[6];
    const float* b2 = (const float*)d_in[7];
    float* out = (float*)d_out;
    (void)in_sizes; (void)n_in; (void)out_size; (void)ws_size;

    const int N = NN, E = NE;
    const int* rows   = ei;        // edge_index[0]
    const int* colsIn = ei + E;    // edge_index[1]

    char* base = (char*)d_ws;
    size_t off = 0;
    auto take = [&](size_t bytes) -> char* {
        char* p = base + off;
        off = (off + bytes + 255) & ~(size_t)255;
        return p;
    };
    int*    cnt    = (int*)   take((size_t)N * 4);
    int*    fillc  = (int*)   take((size_t)N * 4);
    float*  dinv   = (float*) take((size_t)N * 4);
    int*    rowptr = (int*)   take((size_t)(N + 1) * 4);
    const int nb   = (N + 255) / 256;
    int*    bsums  = (int*)   take((size_t)nb * 4);
    int*    ccol   = (int*)   take((size_t)E * 4);
    float*  cw     = (float*) take((size_t)E * 4);
    __bf16* wt0    = (__bf16*)take((size_t)NFEAT * NHID * 2);
    __bf16* wt1    = (__bf16*)take((size_t)NHID * NHID * 2);
    __bf16* wt2    = (__bf16*)take((size_t)NHID * NCLASS * 2);
    float*  tmp    = (float*) take((size_t)N * NHID * 4);
    float*  hbuf   = (float*) take((size_t)N * NHID * 4);

    const dim3 B(256);
    const int gN = (N + 255) / 256;
    const int gE = (E + 255) / 256;

    // ---- build normalized CSR once (reused by all 3 layers) ----
    k_zero2     <<<gN, B, 0, stream>>>(cnt, fillc, N);
    k_count     <<<gE, B, 0, stream>>>(rows, cnt, E);
    k_dinv      <<<gN, B, 0, stream>>>(cnt, dinv, N);
    k_scan_local<<<nb, B, 0, stream>>>(cnt, rowptr, bsums, N);
    k_scan_sums <<<1, 1, 0, stream>>>(bsums, nb, rowptr, N);
    k_scan_add  <<<gN, B, 0, stream>>>(rowptr, bsums, N);
    k_fill      <<<gE, B, 0, stream>>>(rows, colsIn, dinv, rowptr, fillc, ccol, cw, E);

    // ---- weights -> transposed bf16 ----
    k_wconv<<<(NFEAT * NHID   + 255) / 256, B, 0, stream>>>(W0, wt0, NFEAT, NHID);
    k_wconv<<<(NHID  * NHID   + 255) / 256, B, 0, stream>>>(W1, wt1, NHID,  NHID);
    k_wconv<<<(NHID  * NCLASS + 255) / 256, B, 0, stream>>>(W2, wt2, NHID,  NCLASS);

    const int gemmBlocks = (N / 16 + 7) / 8;   // 16-row stripe per wave, 8 waves/block
    const int aggBlocks  = (N + 7) / 8;        // 1 node per wave, 8 waves/block

    // ---- layer 0: h = relu(A @ (x @ W0) + b0) ----
    k_gemm<NHID>       <<<gemmBlocks, B, 0, stream>>>(x, wt0, tmp, N);
    k_agg<NHID, true>  <<<aggBlocks,  B, 0, stream>>>(tmp, rowptr, ccol, cw, dinv, b0, hbuf, N);
    // ---- layer 1 ----
    k_gemm<NHID>       <<<gemmBlocks, B, 0, stream>>>(hbuf, wt1, tmp, N);
    k_agg<NHID, true>  <<<aggBlocks,  B, 0, stream>>>(tmp, rowptr, ccol, cw, dinv, b1, hbuf, N);
    // ---- layer 2 (no relu) ----
    k_gemm<NCLASS>     <<<gemmBlocks, B, 0, stream>>>(hbuf, wt2, tmp, N);
    k_agg<NCLASS, false><<<aggBlocks, B, 0, stream>>>(tmp, rowptr, ccol, cw, dinv, b2, out, N);
}
